// GraphOUNet_82248623718515
// MI455X (gfx1250) — matrix operand order repaired and checked by reading, not compiled, not run on hardware
//
#include <hip/hip_runtime.h>

// ---------------------------------------------------------------------------
// Types for CDNA5 WMMA (gfx1250, wave32)
// ---------------------------------------------------------------------------
typedef __attribute__((ext_vector_type(16))) __bf16 v16bf;
typedef __attribute__((ext_vector_type(8)))  float  v8f;

union Frag { v16bf v; uint4 q[2]; unsigned short s[16]; };

__device__ __forceinline__ unsigned short f2bf(float f) {
    union { float f; unsigned u; } c; c.f = f;
    unsigned r = c.u + 0x7FFFu + ((c.u >> 16) & 1u);   // round-to-nearest-even
    return (unsigned short)(r >> 16);
}

// ---------------------------------------------------------------------------
// Weight repack: f32 [T, Cin, Cout] -> bf16 WMMA B-fragments
//   layout: [t][kb][ct][lane][16]  where K = kb*32 + (lane<16?0:16) + i,
//           col = ct*16 + lane%16, zero-padded for K >= Cin (conv1).
// Per-lane fragment becomes 32 contiguous bytes -> two global_load_b128.
// ---------------------------------------------------------------------------
__global__ void k_pack_w(const float* __restrict__ W, unsigned short* __restrict__ o,
                         int Cin, int Cout, int T, int KB)
{
    const int CT = Cout >> 4;
    const long tot = (long)T * KB * CT * 512;
    long i = (long)blockIdx.x * blockDim.x + threadIdx.x;
    for (; i < tot; i += (long)gridDim.x * blockDim.x) {
        const int e    = (int)(i & 15);
        const int lane = (int)((i >> 4) & 31);
        long blk = i >> 9;
        const int ct = (int)(blk % CT); blk /= CT;
        const int kb = (int)(blk % KB);
        const int t  = (int)(blk / KB);
        const int K   = kb * 32 + ((lane < 16) ? 0 : 16) + e;
        const int col = ct * 16 + (lane & 15);
        const float w = (K < Cin) ? W[((size_t)t * Cin + K) * Cout + col] : 0.f;
        o[i] = f2bf(w);
    }
}

// Zero-pad x [N,4] f32 -> [N,32] bf16 (so all gconvs have Cin % 32 == 0)
__global__ void k_pad_x(const float* __restrict__ x, unsigned short* __restrict__ o,
                        int N)
{
    const long tot = (long)N * 32;
    long i = (long)blockIdx.x * blockDim.x + threadIdx.x;
    for (; i < tot; i += (long)gridDim.x * blockDim.x) {
        const int n = (int)(i >> 5), c = (int)(i & 31);
        o[i] = (c < 4) ? f2bf(x[(size_t)n * 4 + c]) : (unsigned short)0;
    }
}

// ---------------------------------------------------------------------------
// gconv: Y[n, co] = sum_t  X[E[t,n], :CIN] @ W[t, :CIN, co]      (f32 accum)
// X bf16 row-major (CIN % 32 == 0), Wp pre-packed bf16 fragments, Y f32.
// HASE=false -> identity gather (linear layer, branch-free).
// One wave32 = one 16x16 output tile. Loads batched per edge type (<=4 K-blocks
// staged), next edge-type index prefetched under the in-flight loads.
// ---------------------------------------------------------------------------
template<int CIN, bool HASE>
__global__ __launch_bounds__(128)
void k_gconv(const unsigned short* __restrict__ X,
             const unsigned short* __restrict__ Wp,
             const int* __restrict__ E,
             float* __restrict__ Y,
             int Nout, int Cout, int T)
{
    constexpr int KB  = CIN / 32;
    constexpr int KBC = (KB > 4) ? 4 : KB;             // staging chunk (VGPR bound)
    const int lane = threadIdx.x;                      // 0..31
    const int tile = blockIdx.x * 4 + threadIdx.y;     // wave-uniform
    if (tile * 16 >= Nout) return;                     // EXEC stays full inside
    const int ct   = blockIdx.y;
    const int CT   = Cout >> 4;
    const int mA   = lane & 15;
    const int kbA2 = (lane < 16) ? 0 : 32;             // A K-base, in bytes
    const int ncol = lane & 15;

    int rowA = tile * 16 + mA;
    if (rowA >= Nout) rowA = Nout - 1;                 // clamp; stores guarded

    // per-lane pointer into packed weights for this cout-tile
    const unsigned short* wl = Wp + ((size_t)ct * 32 + lane) * 16;
    const size_t wstride = (size_t)CT * 512;           // elements per (t,kb) block
    __builtin_prefetch(wl, 0, 0);                      // global_prefetch_b8

    v8f acc = {0.f, 0.f, 0.f, 0.f, 0.f, 0.f, 0.f, 0.f};

    int idx = HASE ? E[rowA] : rowA;                   // index for t = 0

    for (int t = 0; t < T; ++t) {
        const char* xr = (const char*)(X + (size_t)idx * CIN);
        const unsigned short* wt = wl + (size_t)t * KB * wstride;
#pragma unroll
        for (int g0 = 0; g0 < KB; g0 += KBC) {
            Frag a[KBC], b[KBC];
#pragma unroll
            for (int k = 0; k < KBC; ++k) {            // batch loads (clause)
                const int kb = g0 + k;
                a[k].q[0] = *(const uint4*)(xr + kb * 64 + kbA2);
                a[k].q[1] = *(const uint4*)(xr + kb * 64 + kbA2 + 16);
                const uint4* wq = (const uint4*)(wt + (size_t)kb * wstride);
                b[k].q[0] = wq[0];
                b[k].q[1] = wq[1];
            }
            if (HASE && g0 + KBC >= KB && t + 1 < T)   // prefetch next index
                idx = E[(size_t)(t + 1) * Nout + rowA];
#pragma unroll
            for (int k = 0; k < KBC; ++k)              // drain WMMA chain
                acc = __builtin_amdgcn_wmma_f32_16x16x32_bf16(
                          false, a[k].v, false, b[k].v, (short)0, acc, false, false);
        }
    }

    // C/D 16x16 f32 layout: VGPR j holds M = j + (lane<16?0:8), N = lane%16
    const int mBase = (lane < 16) ? 0 : 8;
#pragma unroll
    for (int j = 0; j < 8; ++j) {
        const int row = tile * 16 + mBase + j;
        if (row < Nout)
            Y[(size_t)row * Cout + ct * 16 + ncol] = acc[j];
    }
}

// ---------------------------------------------------------------------------
// Elementwise / reduction helpers
// ---------------------------------------------------------------------------
__global__ void k_zero(float* p, long n) {
    long i = (long)blockIdx.x * blockDim.x + threadIdx.x;
    for (; i < n; i += (long)gridDim.x * blockDim.x) p[i] = 0.f;
}

__global__ void k_f32_to_bf16(const float* a, unsigned short* o, long n) {
    long i = (long)blockIdx.x * blockDim.x + threadIdx.x;
    for (; i < n; i += (long)gridDim.x * blockDim.x) o[i] = f2bf(a[i]);
}

// Per-channel sum / sumsq: LDS partial reduction, then one atomic per channel.
__global__ void k_bn_stats(const float* __restrict__ X, float* __restrict__ st,
                           int N, int C)
{
    __shared__ float ls[256], lq[256];
    const int c   = threadIdx.x % C;
    const int rpb = 256 / C;
    int r = blockIdx.x * rpb + threadIdx.x / C;
    const int rstride = gridDim.x * rpb;
    float s = 0.f, q = 0.f;
    for (; r < N; r += rstride) {
        float v = X[(size_t)r * C + c];
        s += v; q += v * v;
    }
    ls[threadIdx.x] = s; lq[threadIdx.x] = q;
    __syncthreads();
    if (threadIdx.x < (unsigned)C) {
        float S = 0.f, Q = 0.f;
        for (int j = threadIdx.x; j < 256; j += C) { S += ls[j]; Q += lq[j]; }
        atomicAdd(&st[threadIdx.x], S);
        atomicAdd(&st[C + threadIdx.x], Q);
    }
}

// y = g*(x-m)*rsqrt(var+eps)+b  [+ residual] [ReLU]; writes f32 and/or bf16.
__global__ void k_bn_apply(const float* X, const float* st,
                           const float* g, const float* b,
                           const float* res, float* of32, unsigned short* obf,
                           int N, int C, int relu)
{
    const long tot = (long)N * C;
    long i = (long)blockIdx.x * blockDim.x + threadIdx.x;
    const float invN = 1.f / (float)N;
    for (; i < tot; i += (long)gridDim.x * blockDim.x) {
        const int c = (int)(i % C);
        const float m   = st[c] * invN;
        const float var = st[C + c] * invN - m * m;
        float v = g[c] * (X[i] - m) * __frsqrt_rn(var + 1e-5f) + b[c];
        if (res)  v += res[i];
        if (relu) v = fmaxf(v, 0.f);
        if (of32) of32[i] = v;
        if (obf)  obf[i]  = f2bf(v);
    }
}

__global__ void k_add(const float* a, const float* b2, float* of32,
                      unsigned short* obf, long n)
{
    long i = (long)blockIdx.x * blockDim.x + threadIdx.x;
    for (; i < n; i += (long)gridDim.x * blockDim.x) {
        float v = a[i] + b2[i];
        of32[i] = v;
        obf[i]  = f2bf(v);
    }
}

// segment_sum via f32 atomics (global_atomic_add_f32) + per-node count
__global__ void k_segsum(const float* __restrict__ H, const int* __restrict__ par,
                         float* __restrict__ S, float* __restrict__ cnt,
                         int N, int C)
{
    const long tot = (long)N * C;
    long i = (long)blockIdx.x * blockDim.x + threadIdx.x;
    for (; i < tot; i += (long)gridDim.x * blockDim.x) {
        const int n = (int)(i / C), c = (int)(i % C);
        const int p = par[n];
        atomicAdd(&S[(size_t)p * C + c], H[i]);
        if (c == 0) atomicAdd(&cnt[p], 1.f);
    }
}

__global__ void k_segdiv(const float* S, const float* cnt, unsigned short* o,
                         int M, int C)
{
    const long tot = (long)M * C;
    long i = (long)blockIdx.x * blockDim.x + threadIdx.x;
    for (; i < tot; i += (long)gridDim.x * blockDim.x) {
        const int n = (int)(i / C);
        o[i] = f2bf(S[i] / fmaxf(cnt[n], 1.f));
    }
}

// Final head GEMM: out[n,j] = Hid[n,0:32] @ W2[:,j] + b2[j]   (outC = 2 or 4)
__global__ void k_head_out(const float* __restrict__ Hid,
                           const float* __restrict__ W2,
                           const float* __restrict__ B2,
                           float* __restrict__ out, int N, int outC)
{
    const int n = blockIdx.x * blockDim.x + threadIdx.x;
    if (n >= N) return;
    for (int j = 0; j < outC; ++j) {
        float s = B2[j];
#pragma unroll
        for (int k = 0; k < 32; ++k)
            s += Hid[(size_t)n * 32 + k] * W2[k * outC + j];
        out[(size_t)n * outC + j] = s;
    }
}

// ---------------------------------------------------------------------------
// Host orchestration
// ---------------------------------------------------------------------------
extern "C" void kernel_launch(void* const* d_in, const int* in_sizes, int n_in,
                              void* d_out, int out_size, void* d_ws, size_t ws_size,
                              hipStream_t stream)
{
    struct Lin  { const float *W, *b, *g; };
    struct Blk  { const float *W1, *W2, *b1, *b2, *g1, *g2; };
    struct Head { const float *W1, *W2, *b, *b2, *g; };

    int cur = 0;
    auto F  = [&]() -> const float* { return (const float*)d_in[cur++]; };
    auto Ip = [&]() -> const int*   { return (const int*)d_in[cur++]; };
    auto getLin  = [&]() { Lin  l; l.W = F(); l.b = F(); l.g = F(); return l; };
    auto getBlk  = [&]() { Blk  k; k.W1 = F(); k.W2 = F(); k.b1 = F(); k.b2 = F();
                           k.g1 = F(); k.g2 = F(); return k; };
    auto getHead = [&]() { Head h; h.W1 = F(); h.W2 = F(); h.b = F(); h.b2 = F();
                           h.g = F(); return h; };

    // ---- inputs in JAX pytree (sorted-key) flatten order ----
    Ip(); Ip(); Ip();                                    // align_idx 6,7,8 (identity)
    const int* edges[4];                                 // [0]=d5 .. [3]=d8
    for (int d = 0; d < 4; ++d) edges[d] = Ip();
    const int* par_[3];                                  // keys '6','7','8'
    for (int d = 0; d < 3; ++d) par_[d] = Ip();
    const float* c1W = F(); const float* c1b = F(); const float* c1g = F();
    Blk dec_[3][3];                                      // '6','7','8'
    for (int d = 0; d < 3; ++d) for (int i = 0; i < 3; ++i) dec_[d][i] = getBlk();
    Lin down_[3];                                        // '6','7','8'
    for (int d = 0; d < 3; ++d) down_[d] = getLin();
    Blk enc_[4][3];                                      // '5'..'8'
    for (int d = 0; d < 4; ++d) for (int i = 0; i < 3; ++i) enc_[d][i] = getBlk();
    Head pred_[4]; for (int d = 0; d < 4; ++d) pred_[d] = getHead();   // '5'..'8'
    Head regr_[4]; for (int d = 0; d < 4; ++d) regr_[d] = getHead();
    Lin up_[3];    for (int d = 0; d < 3; ++d) up_[d]   = getLin();    // '6','7','8'
    const int* upI[3]; for (int d = 0; d < 3; ++d) upI[d] = Ip();      // '6','7','8'
    const float* x = F();

    const int N_[4] = {2560, 10000, 40000, 150000};      // d5..d8
    const int C_[4] = {256, 128, 64, 32};                // CE == CD per depth

    // ---- workspace bump allocator ----
    char* wp = (char*)d_ws;
    auto alloc = [&](size_t bytes) {
        void* p = (void*)wp; wp += (bytes + 255) & ~(size_t)255; return p;
    };
    float* stats = (float*)alloc(512 * sizeof(float));
    float* y     = (float*)alloc((size_t)4800000 * sizeof(float));
    float* hid   = (float*)alloc((size_t)4800000 * sizeof(float));
    float* decf  = (float*)alloc((size_t)4800000 * sizeof(float));
    float* S     = (float*)alloc((size_t)1280000 * sizeof(float));
    float* cnt   = (float*)alloc((size_t)40000   * sizeof(float));
    unsigned short* wpk  = (unsigned short*)alloc((size_t)524288  * 2); // packed W
    unsigned short* xbf  = (unsigned short*)alloc((size_t)4800000 * 2); // padded x
    unsigned short* hbf  = (unsigned short*)alloc((size_t)4800000 * 2);
    unsigned short* rbf  = (unsigned short*)alloc((size_t)4800000 * 2);
    unsigned short* dbf  = (unsigned short*)alloc((size_t)4800000 * 2);
    unsigned short* pbf  = (unsigned short*)alloc((size_t)1280000 * 2);
    float* convs[4];
    for (int d = 0; d < 4; ++d)
        convs[d] = (float*)alloc((size_t)N_[d] * C_[d] * sizeof(float));

    // ---- launch helpers ----
    auto zero = [&](float* p, long n) {
        k_zero<<<(unsigned)((n + 255) / 256), 256, 0, stream>>>(p, n);
    };
    // CinReal = real K extent of W (4 for conv1), CinPad = padded K (%32==0)
    auto gconv = [&](const unsigned short* Xb, const float* W, const int* E,
                     float* Yo, int Nout, int CinReal, int CinPad, int Cout, int T) {
        const int KB = CinPad / 32;
        long tot = (long)T * KB * (Cout / 16) * 512;
        k_pack_w<<<(unsigned)((tot + 255) / 256), 256, 0, stream>>>(
            W, wpk, CinReal, Cout, T, KB);
        dim3 g((Nout + 63) / 64, Cout / 16);
        dim3 blk(32, 4);
        if (E) {
            switch (CinPad) {
                case 32:  k_gconv<32 ,true ><<<g, blk, 0, stream>>>(Xb, wpk, E, Yo, Nout, Cout, T); break;
                case 64:  k_gconv<64 ,true ><<<g, blk, 0, stream>>>(Xb, wpk, E, Yo, Nout, Cout, T); break;
                case 128: k_gconv<128,true ><<<g, blk, 0, stream>>>(Xb, wpk, E, Yo, Nout, Cout, T); break;
                default:  k_gconv<256,true ><<<g, blk, 0, stream>>>(Xb, wpk, E, Yo, Nout, Cout, T); break;
            }
        } else {
            switch (CinPad) {
                case 32:  k_gconv<32 ,false><<<g, blk, 0, stream>>>(Xb, wpk, E, Yo, Nout, Cout, T); break;
                case 64:  k_gconv<64 ,false><<<g, blk, 0, stream>>>(Xb, wpk, E, Yo, Nout, Cout, T); break;
                case 128: k_gconv<128,false><<<g, blk, 0, stream>>>(Xb, wpk, E, Yo, Nout, Cout, T); break;
                default:  k_gconv<256,false><<<g, blk, 0, stream>>>(Xb, wpk, E, Yo, Nout, Cout, T); break;
            }
        }
    };
    auto bn = [&](const float* Yin, const float* g, const float* b,
                  const float* res, float* of32, unsigned short* obf,
                  int N, int C, int relu) {
        zero(stats, 2 * C);
        k_bn_stats<<<256, 256, 0, stream>>>(Yin, stats, N, C);
        long tot = (long)N * C;
        k_bn_apply<<<(unsigned)((tot + 255) / 256), 256, 0, stream>>>(
            Yin, stats, g, b, res, of32, obf, N, C, relu);
    };
    auto resblocks = [&](float* hf, unsigned short* hb, const Blk* blks,
                         const int* E, int N, int C) {
        for (int i = 0; i < 3; ++i) {
            gconv(hb, blks[i].W1, E, y, N, C, C, C, 7);
            bn(y, blks[i].g1, blks[i].b1, nullptr, nullptr, rbf, N, C, 1);
            gconv(rbf, blks[i].W2, E, y, N, C, C, C, 7);
            bn(y, blks[i].g2, blks[i].b2, hf, hf, hb, N, C, 1);
        }
    };

    // ---- encoder: conv1 + depth 8 -> 5 ----
    k_pad_x<<<(unsigned)(((long)N_[3] * 32 + 255) / 256), 256, 0, stream>>>(
        x, xbf, N_[3]);
    gconv(xbf, c1W, edges[3], y, N_[3], 4, 32, 32, 7);
    bn(y, c1g, c1b, nullptr, convs[3], hbf, N_[3], 32, 1);
    for (int di = 3; di >= 0; --di) {                    // depth 8..5
        resblocks(convs[di], hbf, enc_[di], edges[di], N_[di], C_[di]);
        if (di > 0) {
            const int Np = N_[di - 1], C = C_[di], Cn = C_[di - 1];
            const int* P = par_[di - 1];                 // parent_idx[str(depth)]
            zero(S, (long)Np * C); zero(cnt, Np);
            long tot = (long)N_[di] * C;
            k_segsum<<<(unsigned)((tot + 255) / 256), 256, 0, stream>>>(
                convs[di], P, S, cnt, N_[di], C);
            long tp = (long)Np * C;
            k_segdiv<<<(unsigned)((tp + 255) / 256), 256, 0, stream>>>(
                S, cnt, pbf, Np, C);
            const Lin& dwn = down_[di - 1];
            gconv(pbf, dwn.W, nullptr, y, Np, C, C, Cn, 1);
            bn(y, dwn.g, dwn.b, nullptr, convs[di - 1], hbf, Np, Cn, 1);
        }
    }

    // ---- output layout: logits d5..d8 then signals d5..d8 ----
    float* out = (float*)d_out;
    float* outL[4]; float* outS[4];
    { size_t off = 0;
      for (int d = 0; d < 4; ++d) { outL[d] = out + off; off += (size_t)N_[d] * 2; }
      for (int d = 0; d < 4; ++d) { outS[d] = out + off; off += (size_t)N_[d] * 4; } }

    auto heads = [&](const unsigned short* db, int N, int C,
                     const Head& P, const Head& R, float* oL, float* oS) {
        gconv(db, P.W1, nullptr, y, N, C, C, 32, 1);
        bn(y, P.g, P.b, nullptr, hid, nullptr, N, 32, 1);
        k_head_out<<<(N + 255) / 256, 256, 0, stream>>>(hid, P.W2, P.b2, oL, N, 2);
        gconv(db, R.W1, nullptr, y, N, C, C, 32, 1);
        bn(y, R.g, R.b, nullptr, hid, nullptr, N, 32, 1);
        k_head_out<<<(N + 255) / 256, 256, 0, stream>>>(hid, R.W2, R.b2, oS, N, 4);
    };

    // ---- decoder: d=5 heads, then d=6,7,8 (up + skip + resblocks + heads) ----
    k_f32_to_bf16<<<(unsigned)(((long)N_[0] * C_[0] + 255) / 256), 256, 0, stream>>>(
        convs[0], dbf, (long)N_[0] * C_[0]);
    heads(dbf, N_[0], C_[0], pred_[0], regr_[0], outL[0], outS[0]);

    for (int di = 1; di < 4; ++di) {
        const int N = N_[di], Cp = C_[di - 1], C = C_[di];
        const Lin& up = up_[di - 1];
        gconv(dbf, up.W, upI[di - 1], y, N, Cp, Cp, C, 1);  // gather-upsample + linear
        bn(y, up.g, up.b, nullptr, decf, nullptr, N, C, 1);
        long tot = (long)N * C;
        k_add<<<(unsigned)((tot + 255) / 256), 256, 0, stream>>>(
            decf, convs[di], decf, dbf, tot);               // + skip (align = identity)
        resblocks(decf, dbf, dec_[di - 1], edges[di], N, C);
        heads(dbf, N, C, pred_[di], regr_[di], outL[di], outS[di]);
    }
}
